// IncoherentLinear_6030134083828
// MI455X (gfx1250) — matrix operand order repaired and checked by reading, not compiled, not run on hardware
//
#include <hip/hip_runtime.h>
#include <math.h>

// ---------------------------------------------------------------------------
// IncoherentLinear for MI455X (gfx1250):
//   out = FWHT_M( (FWHT_N(x*SU)*(1/sqrt(N)/32)) @ W^T * Wscale ) * (1/sqrt(M)) * SV*32
// GEMM (549 GFLOP) runs on v_wmma_f32_16x16x32_bf16 with f32 accumulation.
// LDS staging via GLOBAL_LOAD_ASYNC_TO_LDS_B128 (ASYNCcnt), double-buffered,
// with the last K-stage peeled so the steady-state loop is branch-free.
// ---------------------------------------------------------------------------

typedef __attribute__((ext_vector_type(16))) __bf16 v16bf;
typedef __attribute__((ext_vector_type(8)))  __bf16 v8bf;
typedef __attribute__((ext_vector_type(4)))  __bf16 v4bf;
typedef __attribute__((ext_vector_type(8)))  float  v8f;
typedef int v4i __attribute__((__vector_size__(4 * sizeof(int))));

#define N_DIM 4096
#define M_DIM 4096
#define SCALE_F 32.0f

#if __has_builtin(__builtin_amdgcn_global_load_async_to_lds_b128) && \
    __has_builtin(__builtin_amdgcn_s_wait_asynccnt)
#define HAVE_ASYNC_LDS 1
#else
#define HAVE_ASYNC_LDS 0
#endif

#define AS1 __attribute__((address_space(1)))
#define AS3 __attribute__((address_space(3)))

// ---------------------------------------------------------------------------
// Kernel 1: per-row FWHT of x*SU, scale by 1/sqrt(N)/32, store bf16.
// ---------------------------------------------------------------------------
__global__ __launch_bounds__(256) void k_fwht_in(const float* __restrict__ x,
                                                 const float* __restrict__ SU,
                                                 __bf16* __restrict__ A,
                                                 float c1) {
    __shared__ float s[N_DIM];
    const int tid = threadIdx.x;
    const size_t row = blockIdx.x;

    const float4* xr  = (const float4*)(x + row * (size_t)N_DIM);
    const float4* su4 = (const float4*)SU;
#pragma unroll
    for (int c = 0; c < N_DIM / 4 / 256; ++c) {
        int v = c * 256 + tid;
        float4 a = xr[v];
        float4 b = su4[v];
        s[v * 4 + 0] = a.x * b.x;
        s[v * 4 + 1] = a.y * b.y;
        s[v * 4 + 2] = a.z * b.z;
        s[v * 4 + 3] = a.w * b.w;
    }

    for (int h = 1; h < N_DIM; h <<= 1) {
        __syncthreads();
#pragma unroll 2
        for (int p = tid; p < N_DIM / 2; p += 256) {
            int i = ((p & ~(h - 1)) << 1) | (p & (h - 1));
            float a = s[i];
            float b = s[i + h];
            s[i]     = a + b;
            s[i + h] = a - b;
        }
    }
    __syncthreads();

    __bf16* outp = A + row * (size_t)N_DIM;
#pragma unroll
    for (int c = 0; c < 16; ++c) {
        int v = c * 256 + tid;
        outp[v] = (__bf16)(s[v] * c1);
    }
}

// ---------------------------------------------------------------------------
// Kernel 2: Wb[m][n] = bf16(W[m][n] * Wscale[m])
// ---------------------------------------------------------------------------
__global__ __launch_bounds__(256) void k_prep_w(const float* __restrict__ W,
                                                const float* __restrict__ Wscale,
                                                __bf16* __restrict__ Wb,
                                                int total4) {
    int i = blockIdx.x * 256 + threadIdx.x;
    if (i >= total4) return;
    int m = (i * 4) >> 12;  // / N_DIM (4096)
    float ws = Wscale[m];
    float4 w = ((const float4*)W)[i];
    v4bf o;
    o[0] = (__bf16)(w.x * ws);
    o[1] = (__bf16)(w.y * ws);
    o[2] = (__bf16)(w.z * ws);
    o[3] = (__bf16)(w.w * ws);
    ((v4bf*)Wb)[i] = o;
}

// ---------------------------------------------------------------------------
// Kernel 3: C[t][m] = sum_n A[t][n] * Wb[m][n]   (NT GEMM, K contiguous both)
// Block tile 128(t) x 128(m), 8 waves; each wave 32(t) x 64(m) = 2x4 WMMA
// tiles, K-stage 64 (2 x v_wmma K=32 sub-steps), padded LDS stride 72 bf16.
// ---------------------------------------------------------------------------
constexpr int LDK2  = 72;           // padded K stride (bf16 elems, 144B)
constexpr int TILE  = 128 * LDK2;   // one staging tile (elems)

__device__ __forceinline__ void compute_stage(const __bf16* __restrict__ Ab,
                                              const __bf16* __restrict__ Bb,
                                              int wave_t, int wave_m,
                                              int g, int r, v8f acc[2][4]) {
#pragma unroll
    for (int kk = 0; kk < 64; kk += 32) {
        // A fragment (16x32): lane group g holds K = {g*8..+8, 16+g*8..+8}
        v16bf aF[2];
#pragma unroll
        for (int ti = 0; ti < 2; ++ti) {
            const __bf16* p = Ab + (wave_t * 32 + ti * 16 + r) * LDK2 + kk + g * 8;
            v8bf lo = *(const v8bf*)p;
            v8bf hi = *(const v8bf*)(p + 16);
            aF[ti] = __builtin_shufflevector(lo, hi, 0, 1, 2, 3, 4, 5, 6, 7,
                                             8, 9, 10, 11, 12, 13, 14, 15);
        }
        // B fragment (32x16): lane = column, group g holds K = {g*8..+8, 16+g*8..+8}
        v16bf bF[4];
#pragma unroll
        for (int mj = 0; mj < 4; ++mj) {
            const __bf16* p = Bb + (wave_m * 64 + mj * 16 + r) * LDK2 + kk + g * 8;
            v8bf lo = *(const v8bf*)p;
            v8bf hi = *(const v8bf*)(p + 16);
            bF[mj] = __builtin_shufflevector(lo, hi, 0, 1, 2, 3, 4, 5, 6, 7,
                                             8, 9, 10, 11, 12, 13, 14, 15);
        }
#pragma unroll
        for (int ti = 0; ti < 2; ++ti)
#pragma unroll
            for (int mj = 0; mj < 4; ++mj)
                acc[ti][mj] = __builtin_amdgcn_wmma_f32_16x16x32_bf16(
                    false, aF[ti], false, bF[mj], (short)0, acc[ti][mj],
                    false, false);
    }
}

__global__ __launch_bounds__(256) void k_gemm(const __bf16* __restrict__ A,
                                              const __bf16* __restrict__ B,
                                              float* __restrict__ C) {
#if HAVE_ASYNC_LDS
    __shared__ __align__(16) __bf16 As[2 * TILE];
    __shared__ __align__(16) __bf16 Bs[2 * TILE];
#else
    __shared__ __align__(16) __bf16 As[TILE];
    __shared__ __align__(16) __bf16 Bs[TILE];
#endif

    const int tid    = threadIdx.x;
    const int lane   = tid & 31;
    const int wave   = tid >> 5;
    const int g      = lane >> 4;
    const int r      = lane & 15;
    const int wave_t = wave >> 1;   // 0..3 -> 32-row slice of t
    const int wave_m = wave & 1;    // 0..1 -> 64-col slice of m

    const size_t tb = (size_t)blockIdx.y * 128;
    const size_t mb = (size_t)blockIdx.x * 128;

    v8f acc[2][4] = {};

    // Cooperative 128x64 bf16 tile: 1024 x 16B chunks, 4 per thread per matrix.
    const int r0   = tid >> 3;          // base row, rows r0 + {0,32,64,96}
    const int colK = (tid & 7) * 8;     // K column within 64-wide stage
    const __bf16* gA[4];
    const __bf16* gB[4];
    int ldsOff[4];
#pragma unroll
    for (int i = 0; i < 4; ++i) {
        gA[i] = A + (tb + r0 + i * 32) * (size_t)N_DIM + colK;
        gB[i] = B + (mb + r0 + i * 32) * (size_t)N_DIM + colK;
        ldsOff[i] = (r0 + i * 32) * LDK2 + colK;
    }

#if HAVE_ASYNC_LDS
    // ---- async double-buffered pipeline (GLOBAL_LOAD_ASYNC_TO_LDS_B128) ----
    constexpr int NSTAGES = N_DIM / 64;  // 64
    // Prologue: stage 0 into buffer 0.
#pragma unroll
    for (int i = 0; i < 4; ++i) {
        __builtin_amdgcn_global_load_async_to_lds_b128(
            (AS1 v4i*)gA[i], (AS3 v4i*)(As + ldsOff[i]), 0, 0);
        __builtin_amdgcn_global_load_async_to_lds_b128(
            (AS1 v4i*)gB[i], (AS3 v4i*)(Bs + ldsOff[i]), 0, 0);
    }
    // Steady state: branch-free — issue next stage, wait current, compute.
#pragma unroll 1
    for (int st = 0; st < NSTAGES - 1; ++st) {
        const int cur = st & 1;
        const int nxt = cur ^ 1;
        const int kb  = st * 64;
#pragma unroll
        for (int i = 0; i < 4; ++i) {
            __builtin_amdgcn_global_load_async_to_lds_b128(
                (AS1 v4i*)(gA[i] + kb + 64),
                (AS3 v4i*)(As + nxt * TILE + ldsOff[i]), 0, 0);
            __builtin_amdgcn_global_load_async_to_lds_b128(
                (AS1 v4i*)(gB[i] + kb + 64),
                (AS3 v4i*)(Bs + nxt * TILE + ldsOff[i]), 0, 0);
        }
        __builtin_amdgcn_s_wait_asynccnt(8);  // current stage's 8 complete
        __syncthreads();
        compute_stage(As + cur * TILE, Bs + cur * TILE, wave_t, wave_m, g, r, acc);
        __syncthreads();  // all waves done reading before nxt is refilled
    }
    // Peeled last stage.
    {
        const int cur = (NSTAGES - 1) & 1;
        __builtin_amdgcn_s_wait_asynccnt(0);
        __syncthreads();
        compute_stage(As + cur * TILE, Bs + cur * TILE, wave_t, wave_m, g, r, acc);
    }
#else
    // ---- fallback: register-staged single-buffer pipeline ----
    v8bf rA[4], rB[4];
#pragma unroll
    for (int i = 0; i < 4; ++i) {
        rA[i] = *(const v8bf*)gA[i];
        rB[i] = *(const v8bf*)gB[i];
    }
    for (int kb = 0; kb < N_DIM; kb += 64) {
#pragma unroll
        for (int i = 0; i < 4; ++i) {
            *(v8bf*)(As + ldsOff[i]) = rA[i];
            *(v8bf*)(Bs + ldsOff[i]) = rB[i];
        }
        if (kb + 64 < N_DIM) {
#pragma unroll
            for (int i = 0; i < 4; ++i) {
                rA[i] = *(const v8bf*)(gA[i] + kb + 64);
                rB[i] = *(const v8bf*)(gB[i] + kb + 64);
            }
        }
        __syncthreads();
        compute_stage(As, Bs, wave_t, wave_m, g, r, acc);
        __syncthreads();
    }
#endif

    // Epilogue: D layout -> lane holds column r, rows g*8 + rr (rr = VGPR idx).
#pragma unroll
    for (int ti = 0; ti < 2; ++ti)
#pragma unroll
        for (int mj = 0; mj < 4; ++mj) {
            size_t col  = mb + wave_m * 64 + mj * 16 + r;
            size_t trow = tb + wave_t * 32 + ti * 16 + g * 8;
            float* o = C + trow * (size_t)M_DIM + col;
#pragma unroll
            for (int rr = 0; rr < 8; ++rr)
                o[(size_t)rr * M_DIM] = acc[ti][mj][rr];
        }
}

// ---------------------------------------------------------------------------
// Kernel 4: in-place per-row FWHT on the output + SV * 32 / sqrt(M) scaling.
// ---------------------------------------------------------------------------
__global__ __launch_bounds__(256) void k_fwht_out(float* __restrict__ outp,
                                                  const float* __restrict__ SV,
                                                  float c2) {
    __shared__ float s[M_DIM];
    const int tid = threadIdx.x;
    const size_t row = blockIdx.x;

    float4* o4 = (float4*)(outp + row * (size_t)M_DIM);
#pragma unroll
    for (int c = 0; c < M_DIM / 4 / 256; ++c) {
        int v = c * 256 + tid;
        float4 a = o4[v];
        s[v * 4 + 0] = a.x;
        s[v * 4 + 1] = a.y;
        s[v * 4 + 2] = a.z;
        s[v * 4 + 3] = a.w;
    }

    for (int h = 1; h < M_DIM; h <<= 1) {
        __syncthreads();
#pragma unroll 2
        for (int p = tid; p < M_DIM / 2; p += 256) {
            int i = ((p & ~(h - 1)) << 1) | (p & (h - 1));
            float a = s[i];
            float b = s[i + h];
            s[i]     = a + b;
            s[i + h] = a - b;
        }
    }
    __syncthreads();

    const float4* sv4 = (const float4*)SV;
#pragma unroll
    for (int c = 0; c < M_DIM / 4 / 256; ++c) {
        int v = c * 256 + tid;
        float4 sv = sv4[v];
        float4 rv;
        rv.x = s[v * 4 + 0] * c2 * sv.x;
        rv.y = s[v * 4 + 1] * c2 * sv.y;
        rv.z = s[v * 4 + 2] * c2 * sv.z;
        rv.w = s[v * 4 + 3] * c2 * sv.w;
        o4[v] = rv;
    }
}

// ---------------------------------------------------------------------------
extern "C" void kernel_launch(void* const* d_in, const int* in_sizes, int n_in,
                              void* d_out, int out_size, void* d_ws, size_t ws_size,
                              hipStream_t stream) {
    const float* x      = (const float*)d_in[0];
    const float* W      = (const float*)d_in[1];
    const float* SU     = (const float*)d_in[2];
    const float* SV     = (const float*)d_in[3];
    const float* Wscale = (const float*)d_in[4];

    const int N = in_sizes[2];       // 4096
    const int M = in_sizes[3];       // 4096
    const int T = in_sizes[0] / N;   // 16384 tokens

    // Workspace: A_bf16 (T*N bf16, 128MB) | Wb_bf16 (M*N bf16, 32MB)
    __bf16* Abf = (__bf16*)d_ws;
    __bf16* Wb  = (__bf16*)((char*)d_ws + (size_t)T * N * sizeof(__bf16));
    float*  out = (float*)d_out;

    const float c1 = (1.0f / sqrtf((float)N)) / SCALE_F;
    const float c2 = (1.0f / sqrtf((float)M)) * SCALE_F;

    k_fwht_in<<<T, 256, 0, stream>>>(x, SU, Abf, c1);
    int total4 = (M * N) / 4;
    k_prep_w<<<(total4 + 255) / 256, 256, 0, stream>>>(W, Wscale, Wb, total4);
    k_gemm<<<dim3(M / 128, T / 128), 256, 0, stream>>>(Abf, Wb, out);
    k_fwht_out<<<T, 256, 0, stream>>>(out, SV, c2);
}
